// ResnetRotationPointHeightnet_63788854280406
// MI455X (gfx1250) — matrix operand order repaired and checked by reading, not compiled, not run on hardware
//
#include <hip/hip_runtime.h>
#include <hip/hip_bf16.h>
#include <math.h>

// ---------------------------------------------------------------------------
// Problem constants (from the reference)
// ---------------------------------------------------------------------------
#define B_    16
#define T_    4096
#define HID_  128
#define NCH_  4
#define GRID_ 64
#define HW_   (GRID_ * GRID_)       // 4096 cells
#define L_    (NCH_ + 1)            // 5 planes (last one zero)
#define NIMG_ (B_ * L_)             // 80 unet images
#define FCH_  (HID_ + 3)            // 131 unet input channels
#define CDIM_ 4

typedef __attribute__((ext_vector_type(16))) __bf16 v16bf;
typedef __attribute__((ext_vector_type(8)))  float  v8f;

#define APITCH 36
#define BPITCH 36

// round-to-nearest-even f32 -> bf16 bits
__device__ __forceinline__ unsigned short f2bf_bits(float f) {
  unsigned u = __builtin_bit_cast(unsigned, f);
  unsigned r = 0x7FFFu + ((u >> 16) & 1u);
  return (unsigned short)((u + r) >> 16);
}

// order-preserving float <-> uint mapping (for atomic max/min on f32)
__device__ __forceinline__ unsigned fmap(float f) {
  unsigned u = __builtin_bit_cast(unsigned, f);
  return (u & 0x80000000u) ? ~u : (u | 0x80000000u);
}
__device__ __forceinline__ float funmap(unsigned u) {
  unsigned v = (u & 0x80000000u) ? (u & 0x7FFFFFFFu) : ~u;
  return __builtin_bit_cast(float, v);
}

union FragBF { unsigned int u[8]; v16bf v; };

// A 16x32 bf16 fragment (ISA 7.12.2): lane = half*16+m ; VGPR v<4 -> k=half*8+2v,
// v>=4 -> k=16+half*8+2(v-4). LDS tile is [m][k] with pitch APITCH.
__device__ __forceinline__ void load_afrag(const unsigned short* lA, int mi, int lane, FragBF& f) {
  const int mrow = lane & 15, half = lane >> 4;
  const unsigned short* ap = lA + (mi * 16 + mrow) * APITCH;
#pragma unroll
  for (int v = 0; v < 8; ++v) {
    int kk = (v < 4) ? (half * 8 + v * 2) : (16 + half * 8 + (v - 4) * 2);
    f.u[v] = *(const unsigned int*)(ap + kk);
  }
}
// B 32x16 bf16 fragment: same K order as A: VGPR v<4 -> k=half*8+2v, v>=4 -> k=16+half*8+2(v-4).
// LDS tile is transposed [n][k] with pitch BPITCH.
__device__ __forceinline__ void load_bfrag(const unsigned short* lB, int ni, int lane, FragBF& f) {
  const int half = lane >> 4;
  const unsigned short* bp = lB + (ni * 16 + (lane & 15)) * BPITCH;
#pragma unroll
  for (int v = 0; v < 8; ++v) {
    int kk = (v < 4) ? (half * 8 + v * 2) : (16 + half * 8 + (v - 4) * 2);
    f.u[v] = *(const unsigned int*)(bp + kk);
  }
}

// ---------------------------------------------------------------------------
// WMMA GEMM: C[M,N] = (reluA? relu(A):A)[M,K] * W[K,N] (+bias)(+resid)(relu?)
// Weights come pre-converted to bf16 and transposed: Wt[N][K].
// Block tile 128x64, 8 waves, each wave owns a 32x32 output (2x2 WMMA tiles).
// B tile staged via CDNA5 GLOBAL_LOAD_ASYNC_TO_LDS_B64 (ASYNCcnt).
// ---------------------------------------------------------------------------
__global__ __launch_bounds__(256) void wmma_gemm_kernel(
    const float* __restrict__ A, const unsigned short* __restrict__ Wt,
    const float* __restrict__ bias, const float* __restrict__ resid,
    float* __restrict__ C, int M, int N, int K, int reluA, int reluOut)
{
  __shared__ alignas(16) unsigned short lA[128 * APITCH];
  __shared__ alignas(16) unsigned short lB[64 * BPITCH];
  const int tid = threadIdx.x, lane = tid & 31, wave = tid >> 5;
  const int bm = blockIdx.y * 128, bn = blockIdx.x * 64;
  const int ni0 = (wave & 1) * 2, mi0 = (wave >> 1) * 2;
  v8f acc00 = {}, acc01 = {}, acc10 = {}, acc11 = {};
  const int bn_n = tid >> 2, bchunk = tid & 3;      // async B staging: 64 rows x 4 chunks of 8B
  const unsigned ldsB = (unsigned)(size_t)(&lB[bn_n * BPITCH + bchunk * 4]);
  for (int k0 = 0; k0 < K; k0 += 32) {
    // async-stage B tile: lB[n][k0..k0+15] <- Wt[bn+n][k0..k0+15] (bf16, 8B per lane)
    if (bn + bn_n < N) {
      unsigned long long ga =
          (unsigned long long)(size_t)(Wt + (size_t)(bn + bn_n) * K + k0 + bchunk * 4);
      asm volatile("global_load_async_to_lds_b64 %0, %1, off"
                   :: "v"(ldsB), "v"(ga) : "memory");
    }
    // stage A tile (f32 -> bf16, optional relu)
    for (int i = tid; i < 128 * 32; i += 256) {
      int r = i >> 5, c = i & 31;
      int gm = bm + r, gk = k0 + c;
      float f = (gm < M && gk < K) ? A[(size_t)gm * K + gk] : 0.0f;
      if (reluA) f = fmaxf(f, 0.0f);
      lA[r * APITCH + c] = f2bf_bits(f);
    }
    asm volatile("s_wait_asynccnt 0x0" ::: "memory");
    __syncthreads();
    FragBF a0, a1, b0, b1;
    load_afrag(lA, mi0,     lane, a0);
    load_afrag(lA, mi0 + 1, lane, a1);
    load_bfrag(lB, ni0,     lane, b0);
    load_bfrag(lB, ni0 + 1, lane, b1);
    acc00 = __builtin_amdgcn_wmma_f32_16x16x32_bf16(false, a0.v, false, b0.v, (short)0, acc00, false, false);
    acc01 = __builtin_amdgcn_wmma_f32_16x16x32_bf16(false, a0.v, false, b1.v, (short)0, acc01, false, false);
    acc10 = __builtin_amdgcn_wmma_f32_16x16x32_bf16(false, a1.v, false, b0.v, (short)0, acc10, false, false);
    acc11 = __builtin_amdgcn_wmma_f32_16x16x32_bf16(false, a1.v, false, b1.v, (short)0, acc11, false, false);
    __syncthreads();
  }
#pragma unroll
  for (int tm = 0; tm < 2; ++tm) {
#pragma unroll
    for (int tn = 0; tn < 2; ++tn) {
      v8f acc = tm ? (tn ? acc11 : acc10) : (tn ? acc01 : acc00);
      int nC = bn + (ni0 + tn) * 16 + (lane & 15);
#pragma unroll
      for (int r = 0; r < 8; ++r) {
        int gm = bm + (mi0 + tm) * 16 + (lane >> 4) * 8 + r;
        if (gm < M && nC < N) {
          float val = acc[r];
          if (bias)  val += bias[nC];
          if (resid) val += resid[(size_t)gm * N + nC];
          if (reluOut) val = fmaxf(val, 0.0f);
          C[(size_t)gm * N + nC] = val;
        }
      }
    }
  }
}

// f32 W[K][N] -> bf16 Wt[N][K]
__global__ void convT_kernel(const float* __restrict__ W, unsigned short* __restrict__ Wt,
                             int K, int N) {
  int idx = blockIdx.x * 256 + threadIdx.x;
  if (idx >= K * N) return;
  int k = idx / N, n = idx % N;
  Wt[(size_t)n * K + k] = f2bf_bits(W[idx]);
}

// ---------------------------------------------------------------------------
// Implicit-GEMM WMMA conv: out[img][Cout][H][W], 'SAME' padding.
// Block tile: 64 Cout x 128 pixels, 8 waves, each wave 32x32 (2x2 WMMA tiles).
// K = Cin*KH*KW zero-padded to multiples of 32.
// ---------------------------------------------------------------------------
__global__ __launch_bounds__(256) void wmma_conv_kernel(
    const float* __restrict__ in, const float* __restrict__ wt,
    const float* __restrict__ bias, float* __restrict__ out,
    int Cin, int Cout, int H, int W, int KH, int KW, int relu)
{
  __shared__ alignas(16) unsigned short lA[64 * APITCH];
  __shared__ alignas(16) unsigned short lB[128 * BPITCH];
  const int tid = threadIdx.x, lane = tid & 31, wave = tid >> 5;
  const int img = blockIdx.z;
  const int bm = blockIdx.y * 64, bn = blockIdx.x * 128;
  const int ni0 = (wave & 3) * 2, mi0 = (wave >> 2) * 2;
  const int HWp = H * W, K = Cin * KH * KW;
  const int kh2 = KH >> 1, kw2 = KW >> 1;
  const float* inimg = in + (size_t)img * Cin * HWp;
  v8f acc00 = {}, acc01 = {}, acc10 = {}, acc11 = {};
  for (int k0 = 0; k0 < K; k0 += 32) {
    if (k0 + 32 < K)  // prefetch next input-channel slice
      __builtin_prefetch(inimg + (size_t)((k0 + 32) / (KH * KW)) * HWp, 0, 1);
    for (int i = tid; i < 64 * 32; i += 256) {
      int r = i >> 5, c = i & 31;
      int gm = bm + r, gk = k0 + c;
      float f = (gm < Cout && gk < K) ? wt[(size_t)gm * K + gk] : 0.0f;
      lA[r * APITCH + c] = f2bf_bits(f);
    }
    for (int i = tid; i < 128 * 32; i += 256) {
      int pix = i & 127, kloc = i >> 7;
      int gk = k0 + kloc, pg = bn + pix;
      float f = 0.0f;
      if (gk < K && pg < HWp) {
        int ci = gk / (KH * KW), rs = gk % (KH * KW);
        int y = pg / W + rs / KW - kh2;
        int x = pg % W + rs % KW - kw2;
        if (y >= 0 && y < H && x >= 0 && x < W)
          f = inimg[((size_t)ci * H + y) * W + x];
      }
      lB[pix * BPITCH + kloc] = f2bf_bits(f);
    }
    __syncthreads();
    FragBF a0, a1, b0, b1;
    load_afrag(lA, mi0,     lane, a0);
    load_afrag(lA, mi0 + 1, lane, a1);
    load_bfrag(lB, ni0,     lane, b0);
    load_bfrag(lB, ni0 + 1, lane, b1);
    acc00 = __builtin_amdgcn_wmma_f32_16x16x32_bf16(false, a0.v, false, b0.v, (short)0, acc00, false, false);
    acc01 = __builtin_amdgcn_wmma_f32_16x16x32_bf16(false, a0.v, false, b1.v, (short)0, acc01, false, false);
    acc10 = __builtin_amdgcn_wmma_f32_16x16x32_bf16(false, a1.v, false, b0.v, (short)0, acc10, false, false);
    acc11 = __builtin_amdgcn_wmma_f32_16x16x32_bf16(false, a1.v, false, b1.v, (short)0, acc11, false, false);
    __syncthreads();
  }
#pragma unroll
  for (int tm = 0; tm < 2; ++tm) {
#pragma unroll
    for (int tn = 0; tn < 2; ++tn) {
      v8f acc = tm ? (tn ? acc11 : acc10) : (tn ? acc01 : acc00);
      int pcol = bn + (ni0 + tn) * 16 + (lane & 15);
#pragma unroll
      for (int r = 0; r < 8; ++r) {
        int co = bm + (mi0 + tm) * 16 + (lane >> 4) * 8 + r;
        if (co < Cout && pcol < HWp) {
          float val = acc[r] + (bias ? bias[co] : 0.0f);
          if (relu) val = fmaxf(val, 0.0f);
          out[((size_t)img * Cout + co) * HWp + pcol] = val;
        }
      }
    }
  }
}

// ---------------------------------------------------------------------------
// Glue kernels (bandwidth trivial vs the GEMMs)
// ---------------------------------------------------------------------------
__global__ void fc_pos_kernel(const float* __restrict__ p, const float* __restrict__ w,
                              const float* __restrict__ b, float* __restrict__ out) {
  size_t idx = (size_t)blockIdx.x * 256 + threadIdx.x;
  if (idx >= (size_t)B_ * T_ * 256) return;
  size_t row = idx >> 8; int c = (int)(idx & 255);
  const float* pp = p + row * 3;
  out[idx] = pp[0] * w[c] + pp[1] * w[256 + c] + pp[2] * w[512 + c] + b[c];
}

__global__ void rowmax_kernel(const float* __restrict__ X, float* __restrict__ out, int T, int C) {
  int c = blockIdx.x, b = blockIdx.y;
  __shared__ float red[256];
  float m = -3.4e38f;
  for (int t = threadIdx.x; t < T; t += 256)
    m = fmaxf(m, X[((size_t)b * T + t) * C + c]);
  red[threadIdx.x] = m; __syncthreads();
  for (int s = 128; s > 0; s >>= 1) {
    if (threadIdx.x < s) red[threadIdx.x] = fmaxf(red[threadIdx.x], red[threadIdx.x + s]);
    __syncthreads();
  }
  if (threadIdx.x == 0) out[b * C + c] = red[0];
}

__global__ void concat_kernel(const float* __restrict__ net, const float* __restrict__ pooled,
                              float* __restrict__ X) {
  size_t idx = (size_t)blockIdx.x * 256 + threadIdx.x;
  if (idx >= (size_t)B_ * T_ * 256) return;
  size_t row = idx >> 8; int c = (int)(idx & 255);
  int b = (int)(row / T_);
  X[idx] = (c < 128) ? net[row * 128 + c] : pooled[b * 128 + (c - 128)];
}

// h[b][c] = max_t relu(p . w0[:,c] + b0[c])
__global__ void rot_hmax_kernel(const float* __restrict__ p, const float* __restrict__ w0,
                                const float* __restrict__ b0, float* __restrict__ h) {
  int c = blockIdx.x, b = blockIdx.y;
  __shared__ float red[256];
  float wa = w0[c], wb = w0[128 + c], wc = w0[256 + c], bb = b0[c];
  float m = 0.0f;
  for (int t = threadIdx.x; t < T_; t += 256) {
    const float* pp = p + ((size_t)b * T_ + t) * 3;
    m = fmaxf(m, fmaxf(pp[0] * wa + pp[1] * wb + pp[2] * wc + bb, 0.0f));
  }
  red[threadIdx.x] = m; __syncthreads();
  for (int s = 128; s > 0; s >>= 1) {
    if (threadIdx.x < s) red[threadIdx.x] = fmaxf(red[threadIdx.x], red[threadIdx.x + s]);
    __syncthreads();
  }
  if (threadIdx.x == 0) h[b * 128 + c] = red[0];
}

// tiny dense: out[b][n] = (reluIn? relu(X):X)[b,:K] . W[:,n] + bias (relu?)
__global__ void smalldense_kernel(const float* __restrict__ X, const float* __restrict__ W,
                                  const float* __restrict__ bias, float* __restrict__ out,
                                  int K, int N, int reluIn, int reluOut) {
  __shared__ float xs[128];
  int b = blockIdx.x, tid = threadIdx.x;
  for (int i = tid; i < K; i += blockDim.x) {
    float v = X[(size_t)b * K + i];
    xs[i] = reluIn ? fmaxf(v, 0.0f) : v;
  }
  __syncthreads();
  if (tid < N) {
    float acc = bias ? bias[tid] : 0.0f;
    for (int k = 0; k < K; ++k) acc += xs[k] * W[(size_t)k * N + tid];
    if (reluOut) acc = fmaxf(acc, 0.0f);
    out[(size_t)b * N + tid] = acc;
  }
}

__global__ void bcast_add_kernel(float* __restrict__ net, const float* __restrict__ add) {
  size_t idx = (size_t)blockIdx.x * 256 + threadIdx.x;
  if (idx >= (size_t)B_ * T_ * 128) return;
  size_t row = idx >> 7; int c = (int)(idx & 127);
  int b = (int)(row / T_);
  net[idx] += add[b * 128 + c];
}

__global__ void change_basis_kernel(const float* __restrict__ netpl,
                                    float* __restrict__ cmat, float* __restrict__ outTail) {
  int idx = threadIdx.x;
  if (idx >= B_ * NCH_) return;
  int b = idx / NCH_, l = idx % NCH_;
  const float* np = netpl + b * 36 + l * 9;   // rotm[b,l,:,0] at stride 3
  float nx = np[0], ny = np[3], nz = np[6];
  float inv = rsqrtf(nx * nx + ny * ny + nz * nz);
  nx = nx * inv + 1e-4f; ny = ny * inv + 1e-4f; nz = nz * inv + 1e-4f;
  float v0 = -ny, v1 = nx;                     // v = cross(ez, n), v2 = 0
  float Kx[3][3] = {{0.f, 0.f, v1}, {0.f, 0.f, -v0}, {-v1, v0, 0.f}};
  float K2[3][3];
  for (int i = 0; i < 3; ++i)
    for (int j = 0; j < 3; ++j) {
      float s = 0.f;
      for (int k = 0; k < 3; ++k) s += Kx[i][k] * Kx[k][j];
      K2[i][j] = s;
    }
  float sc = (1.0f - nz) / (v0 * v0 + v1 * v1);
  float R[3][3];
  for (int i = 0; i < 3; ++i)
    for (int j = 0; j < 3; ++j)
      R[i][j] = ((i == j) ? 1.0f : 0.0f) + Kx[i][j] + K2[i][j] * sc;
  float c00 = R[1][1]*R[2][2]-R[1][2]*R[2][1];
  float c01 = R[1][2]*R[2][0]-R[1][0]*R[2][2];
  float c02 = R[1][0]*R[2][1]-R[1][1]*R[2][0];
  float det = R[0][0]*c00 + R[0][1]*c01 + R[0][2]*c02;
  float id = 1.0f / det;
  float CI[3][3];
  CI[0][0]=c00*id; CI[1][0]=c01*id; CI[2][0]=c02*id;
  CI[0][1]=(R[0][2]*R[2][1]-R[0][1]*R[2][2])*id;
  CI[1][1]=(R[0][0]*R[2][2]-R[0][2]*R[2][0])*id;
  CI[2][1]=(R[0][1]*R[2][0]-R[0][0]*R[2][1])*id;
  CI[0][2]=(R[0][1]*R[1][2]-R[0][2]*R[1][1])*id;
  CI[1][2]=(R[0][2]*R[1][0]-R[0][0]*R[1][2])*id;
  CI[2][2]=(R[0][0]*R[1][1]-R[0][1]*R[1][0])*id;
  float bx0=fabsf(R[0][0]), bx1=fabsf(R[1][0]), bx2=fabsf(R[2][0]);
  float by0=fabsf(R[0][1]), by1=fabsf(R[1][1]), by2=fabsf(R[2][1]);
  float cx = (bx0+bx1+bx2) * rsqrtf(bx0*bx0+bx1*bx1+bx2*bx2);
  float cy = (by0+by1+by2) * rsqrtf(by0*by0+by1*by1+by2*by2);
  float nm = fmaxf(cx, cy);
  float o[12] = {CI[0][0],CI[0][1],CI[0][2], CI[1][0],CI[1][1],CI[1][2],
                 CI[2][0],CI[2][1],CI[2][2], nm, nm, nm};
  for (int i = 0; i < 12; ++i) { cmat[idx*12+i] = o[i]; outTail[idx*12+i] = o[i]; }
}

// feat layout: [img=b*5+l][ch(131)][cell(4096)], ch 0..127 net_max, 128 h_min, 129 h_sum->mean, 130 h_max
__global__ void scatter_init_kernel(unsigned* __restrict__ featU, float* __restrict__ featF,
                                    float* __restrict__ cnt) {
  size_t idx = (size_t)blockIdx.x * 256 + threadIdx.x;
  const size_t total1 = (size_t)NIMG_ * FCH_ * HW_;
  const size_t total2 = (size_t)B_ * NCH_ * HW_;
  if (idx < total1) {
    size_t img = idx / ((size_t)FCH_ * HW_);
    size_t ch  = (idx / HW_) % FCH_;
    int l = (int)(img % L_);
    if (l == NCH_ || ch == 129) featF[idx] = 0.0f;   // zero plane / sum channel
    else featU[idx] = 0x80000000u;                    // mapped(0.0f) for max/min
  } else if (idx < total1 + total2) {
    cnt[idx - total1] = 0.0f;
  }
}

__global__ void scatter_kernel(const float* __restrict__ p, const float* __restrict__ net,
                               const float* __restrict__ cmat,
                               unsigned* __restrict__ featU, float* __restrict__ featF,
                               float* __restrict__ cnt) {
  int t = blockIdx.x * 256 + threadIdx.x;
  int l = blockIdx.y, b = blockIdx.z;
  if (t >= T_) return;
  const float* C = cmat + (b * NCH_ + l) * 12;
  const float* pp = p + ((size_t)b * T_ + t) * 3;
  const float is = 1.0f / 0.55f;
  float px = pp[0] * is, py = pp[1] * is, pz = pp[2] * is;
  float scale = 1.0f / (C[9] + 0.05f);
  float q0 = (C[0]*px + C[1]*py + C[2]*pz) * scale;
  float q1 = (C[3]*px + C[4]*py + C[5]*pz) * scale;
  float hgt = (C[6]*px + C[7]*py + C[8]*pz) * scale;
  const float intervalInv = 63.0f / 2.0f;
  float fx = (q0 + 1.0f) * intervalInv;
  float fy = (q1 + 1.0f) * intervalInv;
  fx = (fx >= 63.0f) ? 62.9f : fx; fx = (fx < 0.0f) ? 0.0f : fx;
  fy = (fy >= 63.0f) ? 62.9f : fy; fy = (fy < 0.0f) ? 0.0f : fy;
  int ix = (int)rintf(fx), iy = (int)rintf(fy);
  int cell = ix + GRID_ * iy;
  size_t base = ((size_t)(b * L_ + l) * FCH_) * HW_ + cell;
  const float* nrow = net + ((size_t)b * T_ + t) * 128;
  for (int c = 0; c < 128; ++c)
    atomicMax(&featU[base + (size_t)c * HW_], fmap(nrow[c]));
  atomicMin(&featU[base + (size_t)128 * HW_], fmap(hgt));
  atomicAdd(&featF[base + (size_t)129 * HW_], hgt);
  atomicMax(&featU[base + (size_t)130 * HW_], fmap(hgt));
  atomicAdd(&cnt[(size_t)(b * NCH_ + l) * HW_ + cell], 1.0f);
}

__global__ void scatter_fin_kernel(unsigned* __restrict__ featU, float* __restrict__ featF,
                                   const float* __restrict__ cnt) {
  int idx = blockIdx.x * 256 + threadIdx.x;
  if (idx >= B_ * NCH_ * HW_) return;
  int cell = idx & (HW_ - 1);
  int l = (idx / HW_) % NCH_;
  int b = idx / (HW_ * NCH_);
  size_t base = ((size_t)(b * L_ + l) * FCH_) * HW_ + cell;
  float c = cnt[idx];
  for (int ch = 0; ch < 128; ++ch)
    featF[base + (size_t)ch * HW_] = funmap(featU[base + (size_t)ch * HW_]);
  featF[base + (size_t)128 * HW_] = funmap(featU[base + (size_t)128 * HW_]);
  featF[base + (size_t)129 * HW_] = featF[base + (size_t)129 * HW_] / fmaxf(c, 1.0f);
  featF[base + (size_t)130 * HW_] = funmap(featU[base + (size_t)130 * HW_]);
}

__global__ void pool2_kernel(const float* __restrict__ in, float* __restrict__ out,
                             int C, int H, int W) {
  int Ho = H >> 1, Wo = W >> 1;
  size_t tot = (size_t)NIMG_ * C * Ho * Wo;
  size_t idx = (size_t)blockIdx.x * 256 + threadIdx.x;
  if (idx >= tot) return;
  int x = (int)(idx % Wo); size_t r = idx / Wo;
  int y = (int)(r % Ho); r /= Ho;
  int c = (int)(r % C); int img = (int)(r / C);
  const float* ip = in + (((size_t)img * C + c) * H + 2 * y) * W + 2 * x;
  out[idx] = fmaxf(fmaxf(ip[0], ip[1]), fmaxf(ip[W], ip[W + 1]));
}

__global__ void upcat_kernel(const float* __restrict__ up, const float* __restrict__ skip,
                             float* __restrict__ dst, int C1, int C2, int H, int W) {
  int Ct = C1 + C2;
  size_t tot = (size_t)NIMG_ * Ct * H * W;
  size_t idx = (size_t)blockIdx.x * 256 + threadIdx.x;
  if (idx >= tot) return;
  int x = (int)(idx % W); size_t r = idx / W;
  int y = (int)(r % H); r /= H;
  int c = (int)(r % Ct); int img = (int)(r / Ct);
  if (c < C1)
    dst[idx] = up[(((size_t)img * C1 + c) * (H >> 1) + (y >> 1)) * (W >> 1) + (x >> 1)];
  else
    dst[idx] = skip[(((size_t)img * C2 + (c - C1)) * H + y) * W + x];
}

// NCHW [80][4][4096] -> output (B,L,64,64,CDIM)
__global__ void final_out_kernel(const float* __restrict__ outc, float* __restrict__ out) {
  size_t idx = (size_t)blockIdx.x * 256 + threadIdx.x;
  if (idx >= (size_t)NIMG_ * CDIM_ * HW_) return;
  int c = (int)(idx & 3); size_t r = idx >> 2;
  int pos = (int)(r % HW_); int img = (int)(r / HW_);
  out[idx] = outc[((size_t)img * CDIM_ + c) * HW_ + pos];
}

// ---------------------------------------------------------------------------
// Launch
// ---------------------------------------------------------------------------
extern "C" void kernel_launch(void* const* d_in, const int* in_sizes, int n_in,
                              void* d_out, int out_size, void* d_ws, size_t ws_size,
                              hipStream_t stream) {
  (void)in_sizes; (void)n_in; (void)out_size; (void)ws_size;
  const float* p   = (const float*)d_in[0];
  const float* fcw = (const float*)d_in[1];
  const float* fcb = (const float*)d_in[2];
  // blocks[i]: base 3 + i*5 : w0,b0,w1,b1,ws
  const float* rotw0 = (const float*)d_in[28]; const float* rotb0 = (const float*)d_in[29];
  const float* rotw1 = (const float*)d_in[30]; const float* rotb1 = (const float*)d_in[31];
  const float* rotw2 = (const float*)d_in[32]; const float* rotb2 = (const float*)d_in[33];
  const float* rotw3 = (const float*)d_in[34]; const float* rotb3 = (const float*)d_in[35];
  const float* phw = (const float*)d_in[36];   const float* phb = (const float*)d_in[37];
  const float* e1a_w=(const float*)d_in[38], *e1a_b=(const float*)d_in[39];
  const float* e1b_w=(const float*)d_in[40], *e1b_b=(const float*)d_in[41];
  const float* e2a_w=(const float*)d_in[42], *e2a_b=(const float*)d_in[43];
  const float* e2b_w=(const float*)d_in[44], *e2b_b=(const float*)d_in[45];
  const float* ba_w =(const float*)d_in[46], *ba_b =(const float*)d_in[47];
  const float* bb_w =(const float*)d_in[48], *bb_b =(const float*)d_in[49];
  const float* d2a_w=(const float*)d_in[50], *d2a_b=(const float*)d_in[51];
  const float* d2b_w=(const float*)d_in[52], *d2b_b=(const float*)d_in[53];
  const float* d1a_w=(const float*)d_in[54], *d1a_b=(const float*)d_in[55];
  const float* d1b_w=(const float*)d_in[56], *d1b_b=(const float*)d_in[57];
  const float* out_w=(const float*)d_in[58], *out_b=(const float*)d_in[59];

  float* ws = (float*)d_ws;
  size_t off = 0;
  auto carve = [&](size_t n) { float* q = ws + off; off += n; return q; };
  auto carveU = [&](size_t nU) { return (unsigned short*)carve((nU + 1) / 2); };
  const size_t ROWS = (size_t)B_ * T_;
  float* netX   = carve(ROWS * 256);
  float* t0     = carve(ROWS * 128);
  float* dx     = carve(ROWS * 128);
  float* net    = carve(ROWS * 128);
  float* pooled = carve(B_ * 128);
  float* h1     = carve(B_ * 128);
  float* h2     = carve(B_ * 128);
  float* h3     = carve(B_ * 128);
  float* netpl  = carve(B_ * 64);
  float* cmat   = carve(B_ * NCH_ * 12);
  float* padd   = carve(B_ * 128);
  unsigned short* wt0[5]; unsigned short* wt1[5]; unsigned short* wts[5];
  for (int i = 0; i < 5; ++i) {
    wt0[i] = carveU(128 * 256);
    wt1[i] = carveU(128 * 128);
    wts[i] = carveU(128 * 256);
  }
  float* feat   = carve((size_t)NIMG_ * FCH_ * HW_);
  float* cnt    = carve((size_t)B_ * NCH_ * HW_);
  float* bu_e1  = carve((size_t)NIMG_ * 64 * HW_);
  float* bu_d1  = carve((size_t)NIMG_ * 64 * HW_);
  float* bu_p1  = carve((size_t)NIMG_ * 64 * (HW_ / 4));
  float* bu_e2  = carve((size_t)NIMG_ * 128 * (HW_ / 4));
  float* bu_d2  = carve((size_t)NIMG_ * 128 * (HW_ / 4));
  float* bu_p2  = carve((size_t)NIMG_ * 128 * (HW_ / 16));
  float* bu_ba  = carve((size_t)NIMG_ * 256 * (HW_ / 16));
  float* bu_bt  = carve((size_t)NIMG_ * 256 * (HW_ / 16));
  float* bu_c2  = carve((size_t)NIMG_ * 384 * (HW_ / 4));
  float* bu_t2  = carve((size_t)NIMG_ * 128 * (HW_ / 4));
  float* bu_x2  = carve((size_t)NIMG_ * 128 * (HW_ / 4));
  float* bu_c1  = carve((size_t)NIMG_ * 192 * HW_);
  float* bu_t1  = carve((size_t)NIMG_ * 64 * HW_);
  float* bu_x1  = carve((size_t)NIMG_ * 64 * HW_);
  float* bu_oc  = carve((size_t)NIMG_ * CDIM_ * HW_);

  dim3 blk(256);
  auto nblk = [](size_t n) { return (unsigned)((n + 255) / 256); };

  // ---- pre-convert resblock weights to transposed bf16 (async-stageable) ----
  for (int i = 0; i < 5; ++i) {
    convT_kernel<<<nblk(256 * 128), blk, 0, stream>>>((const float*)d_in[3 + i * 5 + 0], wt0[i], 256, 128);
    convT_kernel<<<nblk(128 * 128), blk, 0, stream>>>((const float*)d_in[3 + i * 5 + 2], wt1[i], 128, 128);
    convT_kernel<<<nblk(256 * 128), blk, 0, stream>>>((const float*)d_in[3 + i * 5 + 4], wts[i], 256, 128);
  }

  // ---- point encoder ----
  fc_pos_kernel<<<nblk(ROWS * 256), blk, 0, stream>>>(p, fcw, fcb, netX);
  for (int i = 0; i < 5; ++i) {
    const float* b0 = (const float*)d_in[3 + i * 5 + 1];
    const float* b1 = (const float*)d_in[3 + i * 5 + 3];
    if (i > 0) {
      rowmax_kernel<<<dim3(128, B_), blk, 0, stream>>>(net, pooled, T_, 128);
      concat_kernel<<<nblk(ROWS * 256), blk, 0, stream>>>(net, pooled, netX);
    }
    dim3 g2(2, (unsigned)(ROWS / 128));
    wmma_gemm_kernel<<<g2, blk, 0, stream>>>(netX, wt0[i], b0, nullptr, t0, (int)ROWS, 128, 256, 1, 0);
    wmma_gemm_kernel<<<g2, blk, 0, stream>>>(t0, wt1[i], b1, nullptr, dx, (int)ROWS, 128, 128, 1, 0);
    wmma_gemm_kernel<<<g2, blk, 0, stream>>>(netX, wts[i], nullptr, dx, net, (int)ROWS, 128, 256, 0, 0);
  }

  // ---- rotation head ----
  rot_hmax_kernel<<<dim3(128, B_), blk, 0, stream>>>(p, rotw0, rotb0, h1);
  smalldense_kernel<<<B_, 128, 0, stream>>>(h1, rotw1, rotb1, h2, 128, 128, 0, 1);
  smalldense_kernel<<<B_, 128, 0, stream>>>(h2, rotw2, rotb2, h3, 128, 128, 0, 1);
  smalldense_kernel<<<B_, 128, 0, stream>>>(h3, rotw3, rotb3, netpl, 128, 36, 0, 0);
  smalldense_kernel<<<B_, 128, 0, stream>>>(netpl, phw, phb, padd, 36, 128, 1, 0);
  bcast_add_kernel<<<nblk(ROWS * 128), blk, 0, stream>>>(net, padd);
  change_basis_kernel<<<1, 64, 0, stream>>>(netpl, cmat,
      (float*)d_out + (size_t)NIMG_ * HW_ * CDIM_);

  // ---- scatter to planes ----
  size_t initTot = (size_t)NIMG_ * FCH_ * HW_ + (size_t)B_ * NCH_ * HW_;
  scatter_init_kernel<<<nblk(initTot), blk, 0, stream>>>((unsigned*)feat, feat, cnt);
  scatter_kernel<<<dim3(T_ / 256, NCH_, B_), blk, 0, stream>>>(p, net, cmat,
      (unsigned*)feat, feat, cnt);
  scatter_fin_kernel<<<nblk((size_t)B_ * NCH_ * HW_), blk, 0, stream>>>((unsigned*)feat, feat, cnt);

  // ---- U-Net (implicit-GEMM WMMA convs) ----
  auto conv = [&](const float* in, int Cin, int Cout, int H, int W, int KH, int KW,
                  const float* w, const float* b, float* out, int relu) {
    dim3 g((unsigned)((H * W + 127) / 128), (unsigned)((Cout + 63) / 64), NIMG_);
    wmma_conv_kernel<<<g, blk, 0, stream>>>(in, w, b, out, Cin, Cout, H, W, KH, KW, relu);
  };
  conv(feat,  131,  64, 64, 64, 3, 3, e1a_w, e1a_b, bu_e1, 1);
  conv(bu_e1,  64,  64, 64, 64, 3, 3, e1b_w, e1b_b, bu_d1, 1);
  pool2_kernel<<<nblk((size_t)NIMG_ * 64 * (HW_ / 4)), blk, 0, stream>>>(bu_d1, bu_p1, 64, 64, 64);
  conv(bu_p1,  64, 128, 32, 32, 3, 3, e2a_w, e2a_b, bu_e2, 1);
  conv(bu_e2, 128, 128, 32, 32, 3, 3, e2b_w, e2b_b, bu_d2, 1);
  pool2_kernel<<<nblk((size_t)NIMG_ * 128 * (HW_ / 16)), blk, 0, stream>>>(bu_d2, bu_p2, 128, 32, 32);
  conv(bu_p2, 128, 256, 16, 16, 3, 3, ba_w, ba_b, bu_ba, 1);
  conv(bu_ba, 256, 256, 16, 16, 3, 3, bb_w, bb_b, bu_bt, 1);
  upcat_kernel<<<nblk((size_t)NIMG_ * 384 * (HW_ / 4)), blk, 0, stream>>>(bu_bt, bu_d2, bu_c2, 256, 128, 32, 32);
  conv(bu_c2, 384, 128, 32, 32, 3, 3, d2a_w, d2a_b, bu_t2, 1);
  conv(bu_t2, 128, 128, 32, 32, 3, 3, d2b_w, d2b_b, bu_x2, 1);
  upcat_kernel<<<nblk((size_t)NIMG_ * 192 * HW_), blk, 0, stream>>>(bu_x2, bu_d1, bu_c1, 128, 64, 64, 64);
  conv(bu_c1, 192,  64, 64, 64, 3, 3, d1a_w, d1a_b, bu_t1, 1);
  conv(bu_t1,  64,  64, 64, 64, 3, 3, d1b_w, d1b_b, bu_x1, 1);
  conv(bu_x1,  64, CDIM_, 64, 64, 1, 1, out_w, out_b, bu_oc, 0);

  final_out_kernel<<<nblk((size_t)NIMG_ * CDIM_ * HW_), blk, 0, stream>>>(bu_oc, (float*)d_out);
}
